// NKF_29240137351422
// MI455X (gfx1250) — matrix operand order repaired and checked by reading, not compiled, not run on hardware
//
#include <hip/hip_runtime.h>
#include <hip/hip_bf16.h>

// ---------------------------------------------------------------- constants
#define SIGLEN 16384
#define NFFT   1024
#define HOP    256
#define BATCH  4
#define NFREQ  513            // NFFT/2+1
#define TSTEPS 65             // SIGLEN/HOP + 1
#define BF     (BATCH * NFREQ)   // 2052
#define LTAPS  4
#define HDIM   80
#define G3     240            // 3*HDIM
#define NT_GRU 15             // 240/16
#define NT_F1  5              // 80/16
#define NT_F2  1              // 4 -> padded 16
#define KT3    3              // K=80 padded to 96 = 3*32
#define FRAGSZ (KT3 * 32 * 16)   // halves per A-matrix fragment buffer (1536)

typedef __attribute__((ext_vector_type(16))) _Float16 v16h;
typedef __attribute__((ext_vector_type(8)))  float    v8f;

// ---------------------------------------------------------------- helpers
__device__ __forceinline__ float2 cmul2(float2 a, float2 b) {
  return make_float2(a.x * b.x - a.y * b.y, a.x * b.y + a.y * b.x);
}
__device__ __forceinline__ float2 cadd2(float2 a, float2 b) {
  return make_float2(a.x + b.x, a.y + b.y);
}
__device__ __forceinline__ float2 csub2(float2 a, float2 b) {
  return make_float2(a.x - b.x, a.y - b.y);
}

__device__ __forceinline__ v8f wmma_f16(v16h a, v16h b, v8f c) {
  // v_wmma_f32_16x16x32_f16: (neg_a, A, neg_b, B, c_mod, C, reuse_a, reuse_b)
  return __builtin_amdgcn_wmma_f32_16x16x32_f16(false, a, false, b, (short)0, c,
                                                false, false);
}

// WMMA 16-bit A-fragment element mapping (16x32 tile):
//   k = kt*32 + 16*(j>>3) + 8*(lane>=16) + (j&7),  m = lane&15
// Inverse (writer side) for element (m, k):
//   lane = m + 16*((k>>3)&1);  j = 8*((k>>4)&1) + (k&7);  kt = k>>5
__device__ __forceinline__ int frag_pos(int m, int k) {
  int lane = m + (((k >> 3) & 1) << 4);
  int j = (((k >> 4) & 1) << 3) + (k & 7);
  return (((k >> 5) * 32) + lane) * 16 + j;
}

// Contiguous per-lane 32B fragment load (ds_load_b128 pair).
__device__ __forceinline__ v16h load_frag(const _Float16* buf, int kt, int lane) {
  return *(const v16h*)(buf + (kt * 32 + lane) * 16);
}

// One wave computes 16x16 output tiles nt0..nt1-1 of A[16x80(f16 frags)] *
// B[80 x 16*NT], B pre-packed as v16h fragments (kt*NT + nt), 32 lanes x 16 h.
template <class EMIT>
__device__ __forceinline__ void wave_gemm(const _Float16* Afrag,
                                          const _Float16* Bpack, int NT,
                                          int nt0, int nt1, int lane,
                                          EMIT&& emit) {
  v16h a0 = load_frag(Afrag, 0, lane);
  v16h a1 = load_frag(Afrag, 1, lane);
  v16h a2 = load_frag(Afrag, 2, lane);
  const v16h* bp = (const v16h*)Bpack;
  for (int nt = nt0; nt < nt1; ++nt) {
    v8f c = {};
    c = wmma_f16(a0, bp[(0 * NT + nt) * 32 + lane], c);
    c = wmma_f16(a1, bp[(1 * NT + nt) * 32 + lane], c);
    c = wmma_f16(a2, bp[(2 * NT + nt) * 32 + lane], c);
    emit(nt, c);
  }
}

// ---------------------------------------------------------------- STFT
// grid (TSTEPS, BATCH, 2), block 256.  Direct windowed DFT per frame.
__global__ __launch_bounds__(256) void stft_kernel(const float* __restrict__ x,
                                                   const float* __restrict__ y,
                                                   float2* __restrict__ Xs,
                                                   float2* __restrict__ Ys) {
  __shared__ float s[NFFT];
  int t = blockIdx.x, b = blockIdx.y;
  const float* sig = blockIdx.z ? y : x;
  float2* out = blockIdx.z ? Ys : Xs;

  for (int n = threadIdx.x; n < NFFT; n += blockDim.x) {
    int j = t * HOP + n - NFFT / 2;     // reflect pad
    if (j < 0) j = -j;
    if (j >= SIGLEN) j = 2 * SIGLEN - 2 - j;
    float w = 0.5f - 0.5f * __cosf(6.28318530717958647692f * (float)n / (float)NFFT);
    s[n] = sig[b * SIGLEN + j] * w;
  }
  __syncthreads();

  for (int f = threadIdx.x; f < NFREQ; f += blockDim.x) {
    float re = 0.f, im = 0.f;
    float w0 = -6.28318530717958647692f * (float)f / (float)NFFT;
    for (int n = 0; n < NFFT; ++n) {
      float sn, cs;
      __sincosf(w0 * (float)n, &sn, &cs);
      re = fmaf(s[n], cs, re);
      im = fmaf(s[n], sn, im);
    }
    out[(size_t)(b * NFREQ + f) * TSTEPS + t] = make_float2(re, im);
  }
}

// ------------------------------------------------- weight fragment packing
// W is [N,K] row-major (PyTorch y = x @ W^T).  B-matrix element B[k][n]=W[n][k].
// Fragment layout: half j of lane l in frag (kt*NT+nt):
//   n = nt*16 + (l&15);  k = kt*32 + 16*(j>>3) + 8*(l>>4) + (j&7)
__global__ __launch_bounds__(256) void pack_b_kernel(const float* __restrict__ W,
                                                     int N, int K, int NT, int KT,
                                                     _Float16* __restrict__ out) {
  int idx = blockIdx.x * blockDim.x + threadIdx.x;
  int total = KT * NT * 512;
  if (idx >= total) return;
  int j = idx & 15;
  int lane = (idx >> 4) & 31;
  int frag = idx >> 9;
  int nt = frag % NT;
  int kt = frag / NT;
  int n = nt * 16 + (lane & 15);
  int k = kt * 32 + ((j >> 3) << 4) + ((lane >> 4) << 3) + (j & 7);
  float v = (n < N && k < K) ? W[n * K + k] : 0.0f;
  out[idx] = (_Float16)v;
}

// ---------------------------------------------------------------- NKF scan
// One block = 128 threads = 4 wave32, owns one 16-row tile for all 65 steps.
__global__ __launch_bounds__(128) void nkf_scan_kernel(
    const float2* __restrict__ Xs, const float2* __restrict__ Ys,
    float2* __restrict__ out,
    const float* __restrict__ fc_wr, const float* __restrict__ fc_br,
    const float* __restrict__ fc_wi, const float* __restrict__ fc_bi,
    const float* __restrict__ prelu_in_p,
    const float* __restrict__ gR_bih, const float* __restrict__ gR_bhh,
    const float* __restrict__ gI_bih, const float* __restrict__ gI_bhh,
    const float* __restrict__ f1_br, const float* __restrict__ f1_bi,
    const float* __restrict__ prelu_out_p,
    const float* __restrict__ f2_br, const float* __restrict__ f2_bi,
    const _Float16* __restrict__ pR_ih, const _Float16* __restrict__ pR_hh,
    const _Float16* __restrict__ pI_ih, const _Float16* __restrict__ pI_hh,
    const _Float16* __restrict__ p1_wr, const _Float16* __restrict__ p1_wi,
    const _Float16* __restrict__ p2_wr, const _Float16* __restrict__ p2_wi,
    size_t packed_bytes) {
  __shared__ float sh_h[4][16][HDIM];                    // f32 hidden (pointwise)
  __shared__ __align__(32) _Float16 fragH[4][FRAGSZ];    // h as WMMA-A frags
  __shared__ __align__(32) _Float16 fragX[2][FRAGSZ];    // f -> rnn -> o frags
  __shared__ float sh_G[16][G3];                         // gates / o_re scratch
  __shared__ float sh_Ghn[16][HDIM];                     // gh_n / o_im scratch
  __shared__ float sh_feat[2][16][9];                    // [xt(4), e, dh(4)]
  __shared__ float sh_hprior[2][16][LTAPS];
  __shared__ float sh_hpost[2][16][LTAPS];
  __shared__ float sh_e[2][16];
  __shared__ float sh_kg[2][16][LTAPS];

  const int tid = threadIdx.x;
  const int lane = tid & 31;
  const int wv = tid >> 5;
  const int base = blockIdx.x * 16;
  const float a_in = prelu_in_p[0];
  const float a_out = prelu_out_p[0];

  // Prefetch packed weights (resident working set) into cache.
  for (size_t off = (size_t)tid * 64; off < packed_bytes; off += 128 * 64)
    __builtin_prefetch(((const char*)pR_ih) + off, 0, 1);

  for (int i = tid; i < 4 * 16 * HDIM; i += 128) (&sh_h[0][0][0])[i] = 0.f;
  // zero frag buffers: K-padding (k>=80) positions must stay zero forever
  for (int i = tid; i < 4 * FRAGSZ; i += 128) (&fragH[0][0])[i] = (_Float16)0.f;
  for (int i = tid; i < 2 * FRAGSZ; i += 128) (&fragX[0][0])[i] = (_Float16)0.f;
  for (int i = tid; i < 2 * 16 * LTAPS; i += 128) {
    (&sh_hprior[0][0][0])[i] = 0.f;
    (&sh_hpost[0][0][0])[i] = 0.f;
  }
  __syncthreads();

  const int nt0 = wv * 4;
  const int nt1 = (wv * 4 + 4 < NT_GRU) ? wv * 4 + 4 : NT_GRU;

  for (int t = 0; t < TSTEPS; ++t) {
    // ---- step begin: taps, dh, prior<-post, innovation e, feature vector
    if (tid < 16) {
      int r = tid;
      int g = base + r;
      if (g >= BF) g = BF - 1;
      float2 xt[LTAPS], hp[LTAPS];
#pragma unroll
      for (int l = 0; l < LTAPS; ++l) {
        int tt = t - (LTAPS - 1) + l;
        xt[l] = (tt >= 0) ? Xs[(size_t)g * TSTEPS + tt] : make_float2(0.f, 0.f);
        float2 prior = make_float2(sh_hprior[0][r][l], sh_hprior[1][r][l]);
        float2 post = make_float2(sh_hpost[0][r][l], sh_hpost[1][r][l]);
        float2 dh = csub2(post, prior);
        hp[l] = post;                       // new prior
        sh_hprior[0][r][l] = post.x;
        sh_hprior[1][r][l] = post.y;
        sh_feat[0][r][l] = xt[l].x;
        sh_feat[1][r][l] = xt[l].y;
        sh_feat[0][r][5 + l] = dh.x;
        sh_feat[1][r][5 + l] = dh.y;
      }
      float2 e = Ys[(size_t)g * TSTEPS + t];
#pragma unroll
      for (int l = 0; l < LTAPS; ++l) e = csub2(e, cmul2(xt[l], hp[l]));
      sh_e[0][r] = e.x;
      sh_e[1][r] = e.y;
      sh_feat[0][r][4] = e.x;
      sh_feat[1][r][4] = e.y;
    }
    __syncthreads();

    // ---- fc_in (9->80 complex dense, K too small for WMMA) + PReLU -> fragX
    for (int q = 0; q < 10; ++q) {
      int u = tid * 10 + q;
      int m = u / HDIM, c = u % HDIM;
      float ar = fc_br[c], ai = fc_bi[c];
#pragma unroll
      for (int k = 0; k < 9; ++k) {
        ar = fmaf(sh_feat[0][m][k], fc_wr[c * 9 + k], ar);
        ai = fmaf(sh_feat[1][m][k], fc_wi[c * 9 + k], ai);
      }
      ar = ar >= 0.f ? ar : a_in * ar;
      ai = ai >= 0.f ? ai : a_in * ai;
      int p = frag_pos(m, c);
      fragX[0][p] = (_Float16)ar;
      fragX[1][p] = (_Float16)ai;
    }
    __syncthreads();

    // ---- 4 GRU cells: (f_re,R)->h_rr, (f_im,R)->h_ir, (f_re,I)->h_ri, (f_im,I)->h_ii
    for (int cell = 0; cell < 4; ++cell) {
      int part = cell & 1;
      const _Float16* Wih = (cell < 2) ? pR_ih : pI_ih;
      const _Float16* Whh = (cell < 2) ? pR_hh : pI_hh;
      const float* bih = (cell < 2) ? gR_bih : gI_bih;
      const float* bhh = (cell < 2) ? gR_bhh : gI_bhh;

      // gi = f @ wih^T + bih   -> sh_G[16][240]
      wave_gemm(fragX[part], Wih, NT_GRU, nt0, nt1, lane,
                [&](int nt, v8f c) {
                  int n0 = lane & 15, mb = (lane >> 4) << 3;
                  int col = nt * 16 + n0;
                  float bv = bih[col];
#pragma unroll
                  for (int rr = 0; rr < 8; ++rr) sh_G[mb + rr][col] = c[rr] + bv;
                });
      __syncthreads();

      // gh = h @ whh^T + bhh ; r,z columns accumulate into G, n-gate -> Ghn
      wave_gemm(fragH[cell], Whh, NT_GRU, nt0, nt1, lane,
                [&](int nt, v8f c) {
                  int n0 = lane & 15, mb = (lane >> 4) << 3;
                  int col = nt * 16 + n0;
                  float bv = bhh[col];
                  if (col < 2 * HDIM) {
#pragma unroll
                    for (int rr = 0; rr < 8; ++rr)
                      sh_G[mb + rr][col] += c[rr] + bv;
                  } else {
#pragma unroll
                    for (int rr = 0; rr < 8; ++rr)
                      sh_Ghn[mb + rr][col - 2 * HDIM] = c[rr] + bv;
                  }
                });
      __syncthreads();

      // pointwise gate math (1280 units / 128 threads); write f32 h + h-frag
      for (int q = 0; q < 10; ++q) {
        int u = tid * 10 + q;
        int m = u / HDIM, j = u % HDIM;
        float rg = 1.f / (1.f + __expf(-sh_G[m][j]));
        float zg = 1.f / (1.f + __expf(-sh_G[m][HDIM + j]));
        float ng = tanhf(fmaf(rg, sh_Ghn[m][j], sh_G[m][2 * HDIM + j]));
        float hv = sh_h[cell][m][j];
        float hn = (1.f - zg) * ng + zg * hv;
        sh_h[cell][m][j] = hn;
        fragH[cell][frag_pos(m, j)] = (_Float16)hn;
      }
      __syncthreads();
    }

    // ---- rnn_out = (h_rr - h_ii) + i(h_ri + h_ir) -> fragX (f frags dead now)
    for (int q = 0; q < 10; ++q) {
      int u = tid * 10 + q;
      int m = u / HDIM, j = u % HDIM;
      float re = sh_h[0][m][j] - sh_h[3][m][j];
      float im = sh_h[2][m][j] + sh_h[1][m][j];
      int p = frag_pos(m, j);
      fragX[0][p] = (_Float16)re;
      fragX[1][p] = (_Float16)im;
    }
    __syncthreads();

    // ---- fco1 (80->80 complex dense): 10 tiles over 4 waves, raw o -> f32
    for (int ttile = wv; ttile < 10; ttile += 4) {
      int part = ttile / 5, nt = ttile % 5;
      const _Float16* Bp = part ? p1_wi : p1_wr;
      const float* bias = part ? f1_bi : f1_br;
      wave_gemm(fragX[part], Bp, NT_F1, nt, nt + 1, lane, [&](int n, v8f c) {
        int n0 = lane & 15, mb = (lane >> 4) << 3;
        int col = n * 16 + n0;
        float bv = bias[col];
#pragma unroll
        for (int rr = 0; rr < 8; ++rr) {
          if (part)
            sh_Ghn[mb + rr][col] = c[rr] + bv;          // o_im (raw)
          else
            sh_G[mb + rr][2 * HDIM + col] = c[rr] + bv; // o_re (raw)
        }
      });
    }
    __syncthreads();

    // ---- PReLU on o -> fragX (rnn frags dead now)
    for (int q = 0; q < 20; ++q) {
      int u = tid * 20 + q;
      int part = u / 1280;
      int v = u % 1280;
      int m = v / HDIM, j = v % HDIM;
      float val = part ? sh_Ghn[m][j] : sh_G[m][2 * HDIM + j];
      val = val >= 0.f ? val : a_out * val;
      fragX[part][frag_pos(m, j)] = (_Float16)val;
    }
    __syncthreads();

    // ---- fco2 (80->4 complex dense): 2 tiles on waves 0,1
    if (wv < 2) {
      int part = wv;
      const _Float16* Bp = part ? p2_wi : p2_wr;
      const float* bias = part ? f2_bi : f2_br;
      wave_gemm(fragX[part], Bp, NT_F2, 0, 1, lane, [&](int n, v8f c) {
        (void)n;
        int n0 = lane & 15, mb = (lane >> 4) << 3;
        if (n0 < LTAPS) {
          float bv = bias[n0];
#pragma unroll
          for (int rr = 0; rr < 8; ++rr) sh_kg[part][mb + rr][n0] = c[rr] + bv;
        }
      });
    }
    __syncthreads();

    // ---- Kalman update, clip, echo, output
    if (tid < 16) {
      int r = tid;
      int g = base + r;
      bool valid = g < BF;
      if (g >= BF) g = BF - 1;
      float2 e = make_float2(sh_e[0][r], sh_e[1][r]);
      float2 echo = make_float2(0.f, 0.f);
#pragma unroll
      for (int l = 0; l < LTAPS; ++l) {
        float2 kg = make_float2(sh_kg[0][r][l], sh_kg[1][r][l]);
        float2 prior = make_float2(sh_hprior[0][r][l], sh_hprior[1][r][l]);
        float2 upd = cadd2(prior, cmul2(kg, e));
        upd.x = fminf(8.f, fmaxf(-8.f, upd.x));
        upd.y = fminf(8.f, fmaxf(-8.f, upd.y));
        sh_hpost[0][r][l] = upd.x;
        sh_hpost[1][r][l] = upd.y;
        float2 xt = make_float2(sh_feat[0][r][l], sh_feat[1][r][l]);
        echo = cadd2(echo, cmul2(xt, upd));
      }
      if (valid) {
        float2 yt = Ys[(size_t)g * TSTEPS + t];
        out[(size_t)g * TSTEPS + t] = make_float2(yt.x - echo.x, yt.y - echo.y);
      }
    }
    __syncthreads();
  }
}

// ---------------------------------------------------------------- launch
extern "C" void kernel_launch(void* const* d_in, const int* in_sizes, int n_in,
                              void* d_out, int out_size, void* d_ws,
                              size_t ws_size, hipStream_t stream) {
  (void)in_sizes; (void)n_in; (void)out_size; (void)ws_size;
  const float* x = (const float*)d_in[0];
  const float* y = (const float*)d_in[1];
  const float* fc_wr = (const float*)d_in[2];
  const float* fc_br = (const float*)d_in[3];
  const float* fc_wi = (const float*)d_in[4];
  const float* fc_bi = (const float*)d_in[5];
  const float* prelu_in = (const float*)d_in[6];
  const float* gR_wih = (const float*)d_in[7];
  const float* gR_whh = (const float*)d_in[8];
  const float* gR_bih = (const float*)d_in[9];
  const float* gR_bhh = (const float*)d_in[10];
  const float* gI_wih = (const float*)d_in[11];
  const float* gI_whh = (const float*)d_in[12];
  const float* gI_bih = (const float*)d_in[13];
  const float* gI_bhh = (const float*)d_in[14];
  const float* f1_wr = (const float*)d_in[15];
  const float* f1_br = (const float*)d_in[16];
  const float* f1_wi = (const float*)d_in[17];
  const float* f1_bi = (const float*)d_in[18];
  const float* prelu_out = (const float*)d_in[19];
  const float* f2_wr = (const float*)d_in[20];
  const float* f2_br = (const float*)d_in[21];
  const float* f2_wi = (const float*)d_in[22];
  const float* f2_bi = (const float*)d_in[23];

  // workspace layout
  const size_t specElems = (size_t)BF * TSTEPS;        // 133,380 complex
  float2* Xs = (float2*)d_ws;
  float2* Ys = Xs + specElems;
  _Float16* pk = (_Float16*)(Ys + specElems);          // 32B aligned
  const size_t sz_g = (size_t)KT3 * NT_GRU * 512;      // 23040 halfs
  const size_t sz_f1 = (size_t)KT3 * NT_F1 * 512;      //  7680 halfs
  const size_t sz_f2 = (size_t)KT3 * NT_F2 * 512;      //  1536 halfs
  _Float16* pR_ih = pk;
  _Float16* pR_hh = pR_ih + sz_g;
  _Float16* pI_ih = pR_hh + sz_g;
  _Float16* pI_hh = pI_ih + sz_g;
  _Float16* p1_wr = pI_hh + sz_g;
  _Float16* p1_wi = p1_wr + sz_f1;
  _Float16* p2_wr = p1_wi + sz_f1;
  _Float16* p2_wi = p2_wr + sz_f2;
  const size_t packed_bytes = (4 * sz_g + 2 * sz_f1 + 2 * sz_f2) * sizeof(_Float16);

  // 1) STFT of x and y
  dim3 gs(TSTEPS, BATCH, 2);
  stft_kernel<<<gs, 256, 0, stream>>>(x, y, Xs, Ys);

  // 2) pack weights into fp16 WMMA-B fragments
  auto pack = [&](const float* W, int N, int K, int NT, _Float16* dst) {
    int total = KT3 * NT * 512;
    pack_b_kernel<<<(total + 255) / 256, 256, 0, stream>>>(W, N, K, NT, KT3, dst);
  };
  pack(gR_wih, G3, HDIM, NT_GRU, pR_ih);
  pack(gR_whh, G3, HDIM, NT_GRU, pR_hh);
  pack(gI_wih, G3, HDIM, NT_GRU, pI_ih);
  pack(gI_whh, G3, HDIM, NT_GRU, pI_hh);
  pack(f1_wr, HDIM, HDIM, NT_F1, p1_wr);
  pack(f1_wi, HDIM, HDIM, NT_F1, p1_wi);
  pack(f2_wr, LTAPS, HDIM, NT_F2, p2_wr);
  pack(f2_wi, LTAPS, HDIM, NT_F2, p2_wi);

  // 3) sequential NKF scan, WMMA GEMMs, one block per 16-row tile
  int nblocks = (BF + 15) / 16;                        // 129
  nkf_scan_kernel<<<nblocks, 128, 0, stream>>>(
      Xs, Ys, (float2*)d_out, fc_wr, fc_br, fc_wi, fc_bi, prelu_in, gR_bih,
      gR_bhh, gI_bih, gI_bhh, f1_br, f1_bi, prelu_out, f2_br, f2_bi, pR_ih,
      pR_hh, pI_ih, pI_hh, p1_wr, p1_wi, p2_wr, p2_wi, packed_bytes);
}
